// AttentionLayer_66889820668334
// MI455X (gfx1250) — matrix operand order repaired and checked by reading, not compiled, not run on hardware
//
#include <hip/hip_runtime.h>

// ---------------------------------------------------------------------------
// Problem constants (from reference): x[B,T,S], mask[B,T,S], W[T,T], b[T]
// out[b,t] = sum_s att[b,s,t] * xp[b,s,t],  xp[b,s,t] = x[b,t,s]*mask[b,t,s]
// att[b,s,:] = e / (sum_u e + EPS),  e = exp(tanh(xp[b,s,:] @ W + bias))
// ---------------------------------------------------------------------------
#define B_DIM 64
#define T_DIM 512
#define S_DIM 1024
#define EPS_F 1e-7f

#define MT 64          // s-rows per workgroup tile (M of the GEMM)
#define KT 32          // WMMA K step (bf16: 16x16x32)
#define NTHREADS 512   // 16 waves of 32
#define XP 516         // LDS pitch for xp tile (bf16 elems, even)
#define KP 34          // LDS pitch for W slab (bf16 elems, even)

typedef __attribute__((ext_vector_type(16))) __bf16 v16bf;
typedef __attribute__((ext_vector_type(8)))  float  v8f;

union V16BF { v16bf v; unsigned int u[8]; };

__device__ __forceinline__ unsigned short f2bf(float f) {
  unsigned int u = __float_as_uint(f);
  u += 0x7FFFu + ((u >> 16) & 1u);          // round-to-nearest-even
  return (unsigned short)(u >> 16);
}
__device__ __forceinline__ float bf2f(unsigned short h) {
  return __uint_as_float(((unsigned int)h) << 16);
}

__global__ void zero_out_kernel(float* __restrict__ out, int n) {
  int i = blockIdx.x * blockDim.x + threadIdx.x;
  if (i < n) out[i] = 0.0f;
}

__global__ __launch_bounds__(NTHREADS)
void attn_fused_kernel(const float* __restrict__ x,
                       const int*   __restrict__ mask,
                       const float* __restrict__ W,
                       const float* __restrict__ bias,
                       float*       __restrict__ out) {
  // xp tile [s=MT][t=T] in bf16 (A-operand AND att*xp multiplicand)
  __shared__ __align__(16) unsigned short xs[MT * XP];        // 66048 B
  // W K-slab transposed: wt[n][k], k in [0,KT)                // 34816 B
  __shared__ __align__(16) unsigned short wt[T_DIM * KP];
  __shared__ float rowsum[MT];

  const int tid  = threadIdx.x;
  const int bI   = blockIdx.y;
  const int s0   = blockIdx.x * MT;

  const int w     = tid >> 5;
  const int lane  = tid & 31;
  const int grp   = lane >> 4;      // lane half (0/1)
  const int l16   = lane & 15;
  const int mBase = (w & 1) * 32;   // this wave's 2 M-tiles: mBase, mBase+16
  const int nBase = (w >> 1) * 64;  // this wave's 4 N-tiles

  if (tid < MT) rowsum[tid] = 0.0f;

  // ---- Stage xp tile: xs[j][t] = bf16(x[b][t][s0+j] * mask[b][t][s0+j]) ----
  {
    const float* xb = x    + (size_t)bI * T_DIM * S_DIM;
    const int*   mb = mask + (size_t)bI * T_DIM * S_DIM;
#pragma unroll
    for (int it = 0; it < (T_DIM * (MT / 4)) / NTHREADS; ++it) {  // 16 iters
      int lin = it * NTHREADS + tid;
      int t   = lin >> 4;             // 16 float4 per t-row
      int j   = (lin & 15) * 4;
      float4 xv = *(const float4*)(xb + (size_t)t * S_DIM + s0 + j);
      int4   mv = *(const int4*)  (mb + (size_t)t * S_DIM + s0 + j);
      xs[(j + 0) * XP + t] = f2bf(xv.x * (float)mv.x);
      xs[(j + 1) * XP + t] = f2bf(xv.y * (float)mv.y);
      xs[(j + 2) * XP + t] = f2bf(xv.z * (float)mv.z);
      xs[(j + 3) * XP + t] = f2bf(xv.w * (float)mv.w);
    }
  }

  float biasv[4];
#pragma unroll
  for (int a = 0; a < 4; ++a) biasv[a] = bias[nBase + a * 16 + l16];

  v8f acc[2][4];
  const v8f vzero = {0.f, 0.f, 0.f, 0.f, 0.f, 0.f, 0.f, 0.f};
#pragma unroll
  for (int mi = 0; mi < 2; ++mi)
#pragma unroll
    for (int a = 0; a < 4; ++a) acc[mi][a] = vzero;

  const unsigned int* xs32 = (const unsigned int*)xs;
  const unsigned int* wt32 = (const unsigned int*)wt;

  // ------------------------------ K loop ------------------------------
  for (int kb = 0; kb < T_DIM; kb += KT) {
    __syncthreads();  // previous slab fully consumed (also covers xp staging)
    if (kb + KT < T_DIM)
      __builtin_prefetch(W + (size_t)(kb + KT) * T_DIM, 0, 1);  // next slab

    // stage W[kb..kb+31][0..511] -> wt[n][k] (bf16, transposed)
#pragma unroll
    for (int it = 0; it < (KT * (T_DIM / 4)) / NTHREADS; ++it) {  // 8 iters
      int lin = it * NTHREADS + tid;
      int kk  = lin >> 7;             // 128 float4 per k-row
      int n   = (lin & 127) * 4;
      float4 wv = *(const float4*)(W + (size_t)(kb + kk) * T_DIM + n);
      wt[(n + 0) * KP + kk] = f2bf(wv.x);
      wt[(n + 1) * KP + kk] = f2bf(wv.y);
      wt[(n + 2) * KP + kk] = f2bf(wv.z);
      wt[(n + 3) * KP + kk] = f2bf(wv.w);
    }
    __syncthreads();

    // A fragments (16x32 bf16): lane = (grp, m=l16); VGPR v holds K pair
    V16BF Afrag[2];
#pragma unroll
    for (int mi = 0; mi < 2; ++mi) {
      int row = mBase + mi * 16 + l16;
#pragma unroll
      for (int v = 0; v < 8; ++v) {
        int K = ((v & 4) ? 16 : 0) + (v & 3) * 2 + grp * 8;
        Afrag[mi].u[v] = xs32[row * (XP / 2) + ((kb + K) >> 1)];
      }
    }
    // B fragments (32x16 bf16): N = l16, K = (v>=4?16:0) + 2*(v&3) + 8*grp
#pragma unroll
    for (int a = 0; a < 4; ++a) {
      V16BF Bfrag;
      int n = nBase + a * 16 + l16;
#pragma unroll
      for (int v = 0; v < 8; ++v) {
        int K = ((v & 4) ? 16 : 0) + (v & 3) * 2 + grp * 8;
        Bfrag.u[v] = wt32[n * (KP / 2) + (K >> 1)];
      }
      acc[0][a] = __builtin_amdgcn_wmma_f32_16x16x32_bf16(
          false, Afrag[0].v, false, Bfrag.v, (short)0, acc[0][a], false, false);
      acc[1][a] = __builtin_amdgcn_wmma_f32_16x16x32_bf16(
          false, Afrag[1].v, false, Bfrag.v, (short)0, acc[1][a], false, false);
    }
  }

  // ---- e = exp(tanh(z + bias)); keep e in the accumulators ----
#pragma unroll
  for (int mi = 0; mi < 2; ++mi)
#pragma unroll
    for (int a = 0; a < 4; ++a)
#pragma unroll
      for (int v = 0; v < 8; ++v)
        acc[mi][a][v] = __expf(tanhf(acc[mi][a][v] + biasv[a]));

  // ---- row sums over u (partial per wave, combined in LDS) ----
#pragma unroll
  for (int mi = 0; mi < 2; ++mi) {
#pragma unroll
    for (int v = 0; v < 8; ++v) {
      float p = acc[mi][0][v] + acc[mi][1][v] + acc[mi][2][v] + acc[mi][3][v];
      p += __shfl_xor(p, 1);
      p += __shfl_xor(p, 2);
      p += __shfl_xor(p, 4);
      p += __shfl_xor(p, 8);
      if (l16 == 0)
        atomicAdd(&rowsum[mBase + mi * 16 + grp * 8 + v], p);
    }
  }
  __syncthreads();

  // ---- out[b,u] += sum_s att[s,u] * xp[s,u] ----
#pragma unroll
  for (int a = 0; a < 4; ++a) {
    int U = nBase + a * 16 + l16;
    float sacc = 0.0f;
#pragma unroll
    for (int mi = 0; mi < 2; ++mi) {
#pragma unroll
      for (int v = 0; v < 8; ++v) {
        int row = mBase + mi * 16 + grp * 8 + v;
        float att = acc[mi][a][v] / (rowsum[row] + EPS_F);
        sacc += att * bf2f(xs[row * XP + U]);
      }
    }
    sacc += __shfl_xor(sacc, 16);
    if (lane < 16)
      atomicAdd(&out[bI * T_DIM + U], sacc);
  }
}

extern "C" void kernel_launch(void* const* d_in, const int* in_sizes, int n_in,
                              void* d_out, int out_size, void* d_ws, size_t ws_size,
                              hipStream_t stream) {
  const float* x    = (const float*)d_in[0];
  const int*   mask = (const int*)d_in[1];
  const float* W    = (const float*)d_in[2];
  const float* bias = (const float*)d_in[3];
  float* out = (float*)d_out;

  const int n_out = B_DIM * T_DIM;
  zero_out_kernel<<<(n_out + 255) / 256, 256, 0, stream>>>(out, n_out);

  dim3 grid(S_DIM / MT, B_DIM);  // 16 x 64 workgroups
  attn_fused_kernel<<<grid, NTHREADS, 0, stream>>>(x, mask, W, bias, out);
}
